// CasualAttention_75969381532455
// MI455X (gfx1250) — matrix operand order repaired and checked by reading, not compiled, not run on hardware
//
#include <hip/hip_runtime.h>

typedef __attribute__((ext_vector_type(16))) _Float16 v16h;
typedef __attribute__((ext_vector_type(8)))  _Float16 v8h;
typedef __attribute__((ext_vector_type(8)))  float    v8f;
typedef __attribute__((ext_vector_type(4)))  int      v4i;

#define B_DIM 4
#define T_DIM 4096
#define D_DIM 256

// 1/sqrt(256) * log2(e): scores land directly in the exp2 domain
#define Q_SCALE 0.09016844f

// ---- CDNA5 async global->LDS path (verified present on this toolchain) ----
#if __has_builtin(__builtin_amdgcn_global_load_async_to_lds_b128)
#define HAVE_ASYNC 1
#else
#define HAVE_ASYNC 0
#endif

#if HAVE_ASYNC
#if __has_builtin(__builtin_amdgcn_s_wait_asynccnt)
#define WAIT_ASYNC(n) __builtin_amdgcn_s_wait_asynccnt(n)
#else
#define WAIT_ASYNC(n) asm volatile("s_wait_asynccnt %0" ::"n"(n) : "memory")
#endif
#else
#define WAIT_ASYNC(n)
#endif

typedef __attribute__((address_space(1))) v4i global_v4i;
typedef __attribute__((address_space(3))) v4i lds_v4i;

static __device__ __forceinline__ void cp16_to_lds(const void* g, void* l) {
#if HAVE_ASYNC
  global_v4i* gp = (global_v4i*)const_cast<void*>(g);
  lds_v4i*    lp = (lds_v4i*)l;
  __builtin_amdgcn_global_load_async_to_lds_b128(gp, lp, 0, 0);
#else
  *(v4i*)l = *(const v4i*)g;
#endif
}

static __device__ __forceinline__ v8f wmma_f16(v16h a, v16h b, v8f c) {
  return __builtin_amdgcn_wmma_f32_16x16x32_f16(
      false, a, false, b, (short)0, c, false, false);
}

// ---------------------------------------------------------------------------
// Kernel 0: one-shot fp32 -> f16 conversion (x and the three W matrices).
// Hoists all cvt work out of the GEMM hot loops.
// ---------------------------------------------------------------------------
__global__ __launch_bounds__(256) void cvt_f32_f16_kernel(
    const float* __restrict__ s, _Float16* __restrict__ d, int n8)
{
  const int i = blockIdx.x * 256 + threadIdx.x;
  if (i < n8) {
    v8f v = *(const v8f*)(s + (size_t)i * 8);
    v8h o;
    #pragma unroll
    for (int j = 0; j < 8; ++j) o[j] = (_Float16)v[j];
    *(v8h*)(d + (size_t)i * 8) = o;
  }
}

// ---------------------------------------------------------------------------
// Kernel 1: fused QKV projection (x16[BT,256] @ W16[256,256] + b), all-f16
// operands -> b128 x loads + per-element f16 W loads feeding WMMAs, no VALU in the k-loop.
// One wave computes a 16x64 output slab (4 independent accumulator chains).
//   z==0: q stored TRANSPOSED [B, D, T], pre-scaled by Q_SCALE (1/sqrt(D)*log2e)
//   z==1: k row-major [B*T, D]     z==2: v row-major [B*T, D]
// ---------------------------------------------------------------------------
__global__ __launch_bounds__(32, 1) void qkv_proj_kernel(
    const _Float16* __restrict__ x16,
    const _Float16* __restrict__ w16,   // [3][256][256] f16 (q,k,v)
    const float* __restrict__ bq, const float* __restrict__ bk,
    const float* __restrict__ bv,
    _Float16* __restrict__ qT, _Float16* __restrict__ kO,
    _Float16* __restrict__ vO)
{
  const int rt   = blockIdx.x;          // 16-row tile over B*T
  const int cg   = blockIdx.y;          // 64-col group over D
  const int z    = blockIdx.z;          // 0=q 1=k 2=v
  const int lane = threadIdx.x;
  const int n    = lane & 15;
  const int hi   = lane >> 4;
  const int koff = hi * 8;

  const _Float16* W = w16 + (size_t)z * D_DIM * D_DIM;
  const float* bias = (z == 0) ? bq : (z == 1) ? bk : bv;

  const int col0 = cg * 64;
  const int row0 = rt * 16;

  v8f acc[4];
  #pragma unroll
  for (int c4 = 0; c4 < 4; ++c4) {
    const float bb = bias[col0 + c4 * 16 + n];
    #pragma unroll
    for (int i = 0; i < 8; ++i) acc[c4][i] = bb;
  }

  const _Float16* xp = x16 + (size_t)(row0 + n) * D_DIM + koff;

  #pragma unroll
  for (int kk = 0; kk < 8; ++kk) {
    const int k0 = kk * 32;
    v8h alo = *(const v8h*)(xp + k0);
    v8h ahi = *(const v8h*)(xp + k0 + 16);
    v16h a;
    #pragma unroll
    for (int i = 0; i < 8; ++i) { a[i] = alo[i]; a[8 + i] = ahi[i]; }

    const _Float16* wrowp = W + (size_t)(k0 + koff) * D_DIM + col0 + n;
    #pragma unroll
    for (int c4 = 0; c4 < 4; ++c4) {
      v16h bm;
      #pragma unroll
      for (int i = 0; i < 8; ++i) {
        bm[i]     = wrowp[(size_t)i * D_DIM + c4 * 16];
        bm[8 + i] = wrowp[(size_t)(16 + i) * D_DIM + c4 * 16];
      }
      acc[c4] = wmma_f16(a, bm, acc[c4]);
    }
  }

  #pragma unroll
  for (int c4 = 0; c4 < 4; ++c4) {
    const int col = col0 + c4 * 16;
    if (z == 0) {
      const int b  = row0 >> 12;
      const int t0 = (row0 & (T_DIM - 1)) + koff;
      v8h o;
      #pragma unroll
      for (int i = 0; i < 8; ++i) o[i] = (_Float16)(acc[c4][i] * Q_SCALE);
      *(v8h*)(qT + (size_t)(b * D_DIM + col + n) * T_DIM + t0) = o;
    } else {
      _Float16* dst = (z == 1) ? kO : vO;
      #pragma unroll
      for (int i = 0; i < 8; ++i)
        dst[(size_t)(row0 + koff + i) * D_DIM + col + n] = (_Float16)acc[c4][i];
    }
  }
}

// ---------------------------------------------------------------------------
// Kernel 2: causal flash attention.
// Block = 256 threads = 8 waves on one WGP, 64 queries per block.
//   wave w: query tile (w&3), key-split (w>>2) of 2.
// K/V 32x256 f16 tiles staged in LDS via global_load_async_to_lds_b128
// (double-buffered, issue-next-then-wait-previous) and shared by the split's
// 4 waves -> 4x less L2 traffic. Rows padded to 528B for bank striping.
// Softmax runs in the exp2 domain (log2e folded into Q's scale).
// Splits merged at the end with (m,l)-weighted flash combine through LDS.
// ---------------------------------------------------------------------------
#define KSTRIDE_H 264                      // 256 + 8 halves padding
#define TILE_B    (32 * KSTRIDE_H * 2)     // 16896 bytes per 32x256 tile
#define STAGE_B   (8 * TILE_B)             // 2 splits * 2 bufs * 2 mats
#define STATS_OFF 65536                    // combine stats (inside STAGE_B)

__global__ __launch_bounds__(256, 2) void causal_attn_kernel(
    const _Float16* __restrict__ qT,
    const _Float16* __restrict__ kM,
    const _Float16* __restrict__ vM,
    float* __restrict__ out)
{
  extern __shared__ char smem_raw[];

  const int blk   = blockIdx.x;            // 64-query block
  const int b     = blockIdx.y;
  const int tid   = threadIdx.x;
  const int wave  = tid >> 5;
  const int lane  = tid & 31;
  const int n     = lane & 15;
  const int hi    = lane >> 4;
  const int koff  = hi * 8;
  const int qt    = wave & 3;              // query tile in block
  const int split = wave >> 2;             // key split (0/1)
  const int q0    = blk * 64 + qt * 16;

  const _Float16* qTb = qT + (size_t)b * D_DIM * T_DIM;
  const _Float16* kB  = kM + (size_t)b * T_DIM * D_DIM;
  const _Float16* vB  = vM + (size_t)b * T_DIM * D_DIM;

  // Q^T B-fragments (loop-invariant)
  v16h qf[8];
  #pragma unroll
  for (int ds = 0; ds < 8; ++ds)
    #pragma unroll
    for (int i = 0; i < 8; ++i) {
      qf[ds][i]     = qTb[(size_t)(ds * 32 + koff + i) * T_DIM + q0 + n];
      qf[ds][8 + i] = qTb[(size_t)(ds * 32 + 16 + koff + i) * T_DIM + q0 + n];
    }

  v8f acc[16];
  #pragma unroll
  for (int t = 0; t < 16; ++t) { v8f zz = {}; acc[t] = zz; }

  float run_max = -3.0e38f;
  float run_sum = 0.0f;

  const int nj = blk + 1;                  // steps per split

  // loader: thread ls of this split copies 128B of K and 128B of V
  const int ls  = (wave & 3) * 32 + lane;  // 0..127 within split
  const int lrow = ls >> 2;                // 0..31
  const int lcb  = (ls & 3) * 128;         // byte col within 512B row

  char* stage = smem_raw;                  // [split][buf][mat] tiles
  auto tile_at = [&](int sp, int buf, int mat) -> char* {
    return stage + (size_t)(((sp * 2 + buf) * 2 + mat)) * TILE_B;
  };

  #define ISSUE_LOADS(J, BUF)                                              \
    {                                                                      \
      const int kb_ = (J) * 64 + split * 32;                               \
      const char* gk_ = (const char*)(kB + (size_t)(kb_ + lrow) * D_DIM);  \
      const char* gv_ = (const char*)(vB + (size_t)(kb_ + lrow) * D_DIM);  \
      char* lk_ = tile_at(split, (BUF), 0) + lrow * (KSTRIDE_H * 2) + lcb; \
      char* lv_ = tile_at(split, (BUF), 1) + lrow * (KSTRIDE_H * 2) + lcb; \
      _Pragma("unroll")                                                    \
      for (int o = 0; o < 8; ++o) {                                        \
        cp16_to_lds(gk_ + lcb + o * 16, lk_ + o * 16);                     \
        cp16_to_lds(gv_ + lcb + o * 16, lv_ + o * 16);                     \
      }                                                                    \
    }

  ISSUE_LOADS(0, 0)

  for (int j = 0; j < nj; ++j) {
    const int buf = j & 1;
    if (j + 1 < nj) {
      ISSUE_LOADS(j + 1, buf ^ 1)
      WAIT_ASYNC(16);                      // 16 just issued -> buf j complete
    } else {
      WAIT_ASYNC(0);
    }
    __syncthreads();                       // tiles for step j visible

    const int kbase = j * 64 + split * 32;
    if (kbase <= q0 + 15) {                // wave-uniform causal skip
      const _Float16* ks = (const _Float16*)tile_at(split, buf, 0);
      const _Float16* vs = (const _Float16*)tile_at(split, buf, 1);

      v8f c0 = {}, c1 = {};
      #pragma unroll
      for (int ds = 0; ds < 8; ++ds) {
        const int d0 = ds * 32;
        const _Float16* kp0 = ks + (size_t)n * KSTRIDE_H + d0 + koff;
        v8h l0 = *(const v8h*)kp0;
        v8h h0 = *(const v8h*)(kp0 + 16);
        v16h a0;
        #pragma unroll
        for (int i = 0; i < 8; ++i) { a0[i] = l0[i]; a0[8 + i] = h0[i]; }
        c0 = wmma_f16(a0, qf[ds], c0);

        const _Float16* kp1 = ks + (size_t)(16 + n) * KSTRIDE_H + d0 + koff;
        v8h l1 = *(const v8h*)kp1;
        v8h h1 = *(const v8h*)(kp1 + 16);
        v16h a1;
        #pragma unroll
        for (int i = 0; i < 8; ++i) { a1[i] = l1[i]; a1[8 + i] = h1[i]; }
        c1 = wmma_f16(a1, qf[ds], c1);
      }

      if (kbase + 31 > q0) {               // diagonal / tail masking
        const int q = q0 + n;
        #pragma unroll
        for (int i = 0; i < 8; ++i) {
          if (kbase + koff + i > q)      c0[i] = -3.0e38f;
          if (kbase + 16 + koff + i > q) c1[i] = -3.0e38f;
        }
      }

      // online softmax in exp2 domain (lane = query column of S^T)
      float m = c0[0];
      #pragma unroll
      for (int i = 0; i < 8; ++i) { m = fmaxf(m, c0[i]); m = fmaxf(m, c1[i]); }
      m = fmaxf(m, __shfl_xor(m, 16, 32));
      const float nm    = fmaxf(run_max, m);
      const float alpha = exp2f(run_max - nm);

      float s = 0.0f;
      v16h pf;
      #pragma unroll
      for (int i = 0; i < 8; ++i) {
        float p0 = exp2f(c0[i] - nm);
        float p1 = exp2f(c1[i] - nm);
        s += p0 + p1;
        pf[i]     = (_Float16)p0;
        pf[8 + i] = (_Float16)p1;
      }
      s += __shfl_xor(s, 16, 32);
      run_sum = run_sum * alpha + s;
      run_max = nm;

      float af[8];
      #pragma unroll
      for (int i = 0; i < 8; ++i) af[i] = __shfl(alpha, koff + i, 32);
      #pragma unroll
      for (int t = 0; t < 16; ++t) {
        #pragma unroll
        for (int i = 0; i < 8; ++i) acc[t][i] *= af[i];
      }

      const _Float16* vrow = vs + (size_t)koff * KSTRIDE_H + n;
      #pragma unroll
      for (int dt = 0; dt < 16; ++dt) {
        v16h vf;
        #pragma unroll
        for (int i = 0; i < 8; ++i) {
          vf[i]     = vrow[(size_t)i * KSTRIDE_H + dt * 16];
          vf[8 + i] = vrow[(size_t)(16 + i) * KSTRIDE_H + dt * 16];
        }
        acc[dt] = wmma_f16(pf, vf, acc[dt]);
      }
    }
    __syncthreads();                       // all readers done before reuse
  }

  // ---- merge the two key-splits (staging LDS is reusable now) ----
  if (split == 1) {
    float* op = (float*)(smem_raw + qt * 16384);
    #pragma unroll
    for (int dt = 0; dt < 16; ++dt)
      #pragma unroll
      for (int i = 0; i < 8; ++i)
        op[(koff + i) * D_DIM + dt * 16 + n] = acc[dt][i];
    float* st = (float*)(smem_raw + STATS_OFF + qt * 128);
    if (lane < 16) { st[n] = run_max; st[16 + n] = run_sum; }
  }
  __syncthreads();

  if (split == 0) {
    const float* st = (const float*)(smem_raw + STATS_OFF + qt * 128);
    const float m1 = st[n];
    const float l1 = st[16 + n];
    const float M   = fmaxf(run_max, m1);
    const float w0  = exp2f(run_max - M);
    const float w1  = exp2f(m1 - M);
    const float inv = 1.0f / (run_sum * w0 + l1 * w1);
    const float a0s = w0 * inv;
    const float a1s = w1 * inv;
    float f0[8], f1[8];
    #pragma unroll
    for (int i = 0; i < 8; ++i) {
      f0[i] = __shfl(a0s, koff + i, 32);
      f1[i] = __shfl(a1s, koff + i, 32);
    }
    const float* op = (const float*)(smem_raw + qt * 16384);
    float* ob = out + ((size_t)(b * T_DIM + q0)) * D_DIM + n;
    #pragma unroll
    for (int dt = 0; dt < 16; ++dt) {
      #pragma unroll
      for (int i = 0; i < 8; ++i) {
        const float o1 = op[(koff + i) * D_DIM + dt * 16 + n];
        ob[(size_t)(koff + i) * D_DIM + dt * 16] =
            acc[dt][i] * f0[i] + o1 * f1[i];
      }
    }
  }
}

// ---------------------------------------------------------------------------
extern "C" void kernel_launch(void* const* d_in, const int* in_sizes, int n_in,
                              void* d_out, int out_size, void* d_ws,
                              size_t ws_size, hipStream_t stream) {
  (void)in_sizes; (void)n_in; (void)out_size; (void)ws_size;
  const float* x  = (const float*)d_in[0];
  const float* Wq = (const float*)d_in[1];
  const float* bq = (const float*)d_in[2];
  const float* Wk = (const float*)d_in[3];
  const float* bk = (const float*)d_in[4];
  const float* Wv = (const float*)d_in[5];
  const float* bv = (const float*)d_in[6];
  float* out = (float*)d_out;

  // workspace layout (f16 elements):
  //   qT | kO | vO : B*T*D each (24 MB total)
  //   x16          : B*T*D      (8 MB)
  //   w16          : 3*D*D      (0.4 MB)
  _Float16* ws = (_Float16*)d_ws;
  const size_t mat = (size_t)B_DIM * T_DIM * D_DIM;
  _Float16* qT  = ws;
  _Float16* kO  = ws + mat;
  _Float16* vO  = ws + 2 * mat;
  _Float16* x16 = ws + 3 * mat;
  _Float16* w16 = ws + 4 * mat;

  // one-shot conversions
  cvt_f32_f16_kernel<<<dim3((int)(mat / 8 / 256)), dim3(256), 0, stream>>>(
      x, x16, (int)(mat / 8));
  const int wn8 = D_DIM * D_DIM / 8;       // 8192
  cvt_f32_f16_kernel<<<dim3(wn8 / 256), dim3(256), 0, stream>>>(
      Wq, w16 + 0 * D_DIM * D_DIM, wn8);
  cvt_f32_f16_kernel<<<dim3(wn8 / 256), dim3(256), 0, stream>>>(
      Wk, w16 + 1 * D_DIM * D_DIM, wn8);
  cvt_f32_f16_kernel<<<dim3(wn8 / 256), dim3(256), 0, stream>>>(
      Wv, w16 + 2 * D_DIM * D_DIM, wn8);

  dim3 g1((B_DIM * T_DIM) / 16, D_DIM / 64, 3);
  qkv_proj_kernel<<<g1, dim3(32), 0, stream>>>(x16, w16, bq, bk, bv,
                                               qT, kO, vO);

  dim3 g2(T_DIM / 64, B_DIM);
  causal_attn_kernel<<<g2, dim3(256), STAGE_B, stream>>>(qT, kO, vO, out);
}
